// PointNet2MSGCls_28578712387755
// MI455X (gfx1250) — matrix operand order repaired and checked by reading, not compiled, not run on hardware
//
#include <hip/hip_runtime.h>
#include <hip/hip_bf16.h>

typedef __attribute__((ext_vector_type(16))) __bf16 v16bf;
typedef __attribute__((ext_vector_type(8)))  __bf16 v8bf;
typedef __attribute__((ext_vector_type(8)))  float  v8f;

__device__ inline v16bf cat16(v8bf lo, v8bf hi) {
    return __builtin_shufflevector(lo, hi, 0,1,2,3,4,5,6,7,8,9,10,11,12,13,14,15);
}

// ---------------------------------------------------------------------------
// Weight repack: fp32 (cout x cin) -> bf16 (cout x cinp), K zero-padded to 32.
// ---------------------------------------------------------------------------
__global__ void pack_w_kernel(const float* __restrict__ w, __bf16* __restrict__ out,
                              int cout, int cin, int cinp) {
    int id = blockIdx.x * blockDim.x + threadIdx.x;
    if (id >= cout * cinp) return;
    int r = id / cinp, c = id % cinp;
    out[id] = (c < cin) ? (__bf16)w[r * cin + c] : (__bf16)0.0f;
}

// ---------------------------------------------------------------------------
// Farthest point sampling: one block per batch, sequential npoint iterations,
// parallel min-update + argmax reduction (first-index tie-break like argmax).
// ---------------------------------------------------------------------------
__global__ __launch_bounds__(256) void fps_kernel(const float* __restrict__ xyz,
                                                  int bstride, int istride,
                                                  int n, int npoint, int* __restrict__ out) {
    __shared__ float dist[4096];
    __shared__ float rv[256];
    __shared__ int   ri[256];
    const int b = blockIdx.x, t = threadIdx.x;
    const float* base = xyz + (size_t)b * bstride;
    for (int i = t; i < n; i += 256) dist[i] = 1e10f;
    __syncthreads();
    int far = 0;
    for (int it = 0; it < npoint; ++it) {
        if (t == 0) out[b * npoint + it] = far;
        const float fx = base[far * istride + 0];
        const float fy = base[far * istride + 1];
        const float fz = base[far * istride + 2];
        float bv = -1.0f; int bi = 0;
        for (int i = t; i < n; i += 256) {
            float dx = base[i * istride + 0] - fx;
            float dy = base[i * istride + 1] - fy;
            float dz = base[i * istride + 2] - fz;
            float d  = dx * dx + dy * dy + dz * dz;
            float nd = fminf(dist[i], d);
            dist[i] = nd;
            if (nd > bv) { bv = nd; bi = i; }
        }
        rv[t] = bv; ri[t] = bi;
        __syncthreads();
        for (int s = 128; s > 0; s >>= 1) {
            if (t < s) {
                if (rv[t + s] > rv[t] || (rv[t + s] == rv[t] && ri[t + s] < ri[t])) {
                    rv[t] = rv[t + s]; ri[t] = ri[t + s];
                }
            }
            __syncthreads();
        }
        far = ri[0];
        __syncthreads();
    }
}

__global__ void gather_xyz_kernel(const float* __restrict__ xyz, int bstride, int istride,
                                  const int* __restrict__ fidx, int npoint,
                                  float* __restrict__ out, int total) {
    int id = blockIdx.x * blockDim.x + threadIdx.x;
    if (id >= total) return;
    int c = id % 3, s = (id / 3) % npoint, b = id / (3 * npoint);
    out[id] = xyz[(size_t)b * bstride + (size_t)fidx[b * npoint + s] * istride + c];
}

// ---------------------------------------------------------------------------
// Ball query: one wave32 per centroid. Ballot + prefix compaction gives the
// first `nsample` in-radius indices in ascending order (matches sorted cand).
// ---------------------------------------------------------------------------
__global__ __launch_bounds__(32) void ballquery_kernel(const float* __restrict__ xyz,
                                                       int bstride, int istride, int n,
                                                       const float* __restrict__ centers,
                                                       int S, float r2, int nsample,
                                                       int* __restrict__ out) {
    const int g = blockIdx.x;
    const int b = g / S;
    const int lane = threadIdx.x;
    const float cx = centers[g * 3 + 0];
    const float cy = centers[g * 3 + 1];
    const float cz = centers[g * 3 + 2];
    const float* base = xyz + (size_t)b * bstride;
    int cnt = 0, firstidx = -1;
    for (int s0 = 0; s0 < n; s0 += 32) {
        if (cnt >= nsample) break;
        int i = s0 + lane;
        bool in = false;
        if (i < n) {
            float dx = cx - base[i * istride + 0];
            float dy = cy - base[i * istride + 1];
            float dz = cz - base[i * istride + 2];
            in = (dx * dx + dy * dy + dz * dz) < r2;
        }
        unsigned mask = (unsigned)__ballot(in);
        int pre = __popc(mask & ((1u << lane) - 1u));
        if (in && (cnt + pre) < nsample) out[(size_t)g * nsample + cnt + pre] = i;
        int m = __popc(mask);
        if (firstidx < 0 && m != 0) firstidx = s0 + __ffs(mask) - 1;
        cnt += m;
    }
    if (cnt > nsample) cnt = nsample;
    if (firstidx < 0) firstidx = 0;   // no candidates -> reference maps to 0
    for (int k = cnt + lane; k < nsample; k += 32) out[(size_t)g * nsample + k] = firstidx;
}

// ---------------------------------------------------------------------------
// WMMA group MLP. Activations live in LDS sample-major: X[s][c], ld = C+16
// (pad rotates LDS banks and keeps 16/32-byte alignment). B fragments and D
// writebacks are contiguous 128-bit LDS accesses; A fragments are contiguous
// 128-bit global loads from L2-resident packed weights. Two N-adjacent tiles
// per wave share an A fragment and give independent WMMA chains to hide the
// bf16 WMMA data hazards (4-5 co-exec slots).
// ---------------------------------------------------------------------------
struct GroupArgs {
    const float* xyz;   int xyz_bstride; int xyz_istride;
    const float* center;                                   // [B*S*3] or null
    const float* feat;  int feat_bstride; int feat_istride; int feat_cstride; int nfeat;
    const int*   idx;                                      // [B*S*NS] or null (identity)
    const __bf16 *w1, *w2, *w3;
    const float  *b1, *b2, *b3;
    float* out; int out_bstride; int out_sstride;          // point-major output
    int S;
};

template<int COUT, int NS, bool FINAL>
__device__ inline void layer_epilogue(v8f acc, const float* __restrict__ bias,
                                      __bf16* __restrict__ Y, float* __restrict__ smax,
                                      int tm, int tn, int hlf, int col) {
    constexpr int LDY = COUT + 16;
    const float4 b0 = *(const float4*)(bias + tm * 16 + hlf * 8);
    const float4 b1 = *(const float4*)(bias + tm * 16 + hlf * 8 + 4);
    const float bb[8] = {b0.x, b0.y, b0.z, b0.w, b1.x, b1.y, b1.z, b1.w};
    if constexpr (!FINAL) {
        v8bf yv;                       // lane's 8 acc = 8 consecutive channels
        #pragma unroll
        for (int v = 0; v < 8; ++v)
            yv[v] = (__bf16)fmaxf(acc[v] + bb[v], 0.0f);
        *(v8bf*)(Y + (size_t)(tn * 16 + col) * LDY + tm * 16 + hlf * 8) = yv;
    } else {
        #pragma unroll
        for (int v = 0; v < 8; ++v) {
            float val = fmaxf(acc[v] + bb[v], 0.0f);
            #pragma unroll
            for (int off = 8; off >= 1; off >>= 1)
                val = fmaxf(val, __shfl_xor(val, off, 32));
            if (col == 0)
                atomicMax((int*)&smax[tm * 16 + hlf * 8 + v], __float_as_int(val));
        }
    }
}

template<int CIN, int COUT, int NS, bool FINAL>
__device__ inline void mlp_layer(const __bf16* __restrict__ W, const float* __restrict__ bias,
                                 const __bf16* __restrict__ X, __bf16* __restrict__ Y,
                                 float* __restrict__ smax, int lane, int wave) {
    constexpr int TM = COUT / 16, TN = NS / 16;
    constexpr int LDX = CIN + 16;
    const int hlf = lane >> 4;
    const int col = lane & 15;
    if constexpr ((TN % 2 == 0) && (TM * TN >= 8)) {
        // Paired path: two N-adjacent tiles, shared A fragment, dual acc chains.
        constexpr int TNP = TN / 2, NP = TM * TNP;
        for (int tp = wave; tp < NP; tp += 4) {
            const int tm = tp / TNP, tn0 = (tp % TNP) * 2;
            v8f acc0 = {}, acc1 = {};
            const __bf16* wrow  = W + ((size_t)(tm * 16 + col)) * CIN;
            const __bf16* xrow0 = X + ((size_t)(tn0 * 16 + col)) * LDX;
            const __bf16* xrow1 = xrow0 + (size_t)16 * LDX;
            for (int k0 = 0; k0 < CIN; k0 += 32) {
                v16bf a  = *(const v16bf*)(wrow + k0 + hlf * 16);
                v16bf bm0 = *(const v16bf*)(xrow0 + k0 + hlf * 16);
                v16bf bm1 = *(const v16bf*)(xrow1 + k0 + hlf * 16);
                acc0 = __builtin_amdgcn_wmma_f32_16x16x32_bf16(
                    false, a, false, bm0, (short)0, acc0, false, false);
                acc1 = __builtin_amdgcn_wmma_f32_16x16x32_bf16(
                    false, a, false, bm1, (short)0, acc1, false, false);
            }
            layer_epilogue<COUT, NS, FINAL>(acc0, bias, Y, smax, tm, tn0,     hlf, col);
            layer_epilogue<COUT, NS, FINAL>(acc1, bias, Y, smax, tm, tn0 + 1, hlf, col);
        }
    } else {
        for (int t = wave; t < TM * TN; t += 4) {
            const int tm = t / TN, tn = t % TN;
            v8f acc = {};
            const __bf16* wrow = W + ((size_t)(tm * 16 + col)) * CIN;
            const __bf16* xrow = X + ((size_t)(tn * 16 + col)) * LDX;
            for (int k0 = 0; k0 < CIN; k0 += 32) {
                v16bf a  = *(const v16bf*)(wrow + k0 + hlf * 16);
                v16bf bm = *(const v16bf*)(xrow + k0 + hlf * 16);
                acc = __builtin_amdgcn_wmma_f32_16x16x32_bf16(
                    false, a, false, bm, (short)0, acc, false, false);
            }
            layer_epilogue<COUT, NS, FINAL>(acc, bias, Y, smax, tm, tn, hlf, col);
        }
    }
}

template<int CIN, int C1, int C2, int C3, int NS>
__global__ __launch_bounds__(128) void group_mlp_kernel(GroupArgs A) {
    extern __shared__ char smem_raw[];
    constexpr int CB0  = (CIN > C2) ? CIN : C2;
    constexpr int B0E  = NS * (CB0 + 16);     // X0 (ld CIN+16), later Y2 (ld C2+16)
    constexpr int B1E  = NS * (C1 + 16);      // Y1 (ld C1+16)
    __bf16* buf0 = (__bf16*)smem_raw;
    __bf16* buf1 = buf0 + B0E;
    float*  smax = (float*)(buf1 + B1E);      // C3 channel maxima

    const int g = blockIdx.x;
    const int b = g / A.S, s = g % A.S;
    const int t = threadIdx.x, lane = t & 31, wave = t >> 5;

    for (int c = t; c < C3; c += 128) smax[c] = 0.0f;

    // Gather X0 sample-major: buf0[k*(CIN+16) + r]; rows 0..2 = centered xyz,
    // 3..3+nfeat = gathered feats (contiguous in memory -> coalesced), rest 0.
    const int* gidx = A.idx ? (A.idx + (size_t)g * NS) : nullptr;
    float cc[3] = {0.f, 0.f, 0.f};
    if (A.center) { cc[0] = A.center[g*3+0]; cc[1] = A.center[g*3+1]; cc[2] = A.center[g*3+2]; }
    for (int e = t; e < NS * CIN; e += 128) {
        const int k = e / CIN, r = e % CIN;
        const int p = gidx ? gidx[k] : k;
        float v = 0.0f;
        if (r < 3) {
            v = A.xyz[(size_t)b * A.xyz_bstride + (size_t)p * A.xyz_istride + r] - cc[r];
        } else if (r < 3 + A.nfeat) {
            v = A.feat[(size_t)b * A.feat_bstride + (size_t)p * A.feat_istride
                       + (size_t)(r - 3) * A.feat_cstride];
        }
        buf0[(size_t)k * (CIN + 16) + r] = (__bf16)v;
    }
    __syncthreads();

    mlp_layer<CIN, C1, NS, false>(A.w1, A.b1, buf0, buf1, nullptr, lane, wave);
    __syncthreads();
    mlp_layer<C1, C2, NS, false>(A.w2, A.b2, buf1, buf0, nullptr, lane, wave);
    __syncthreads();
    mlp_layer<C2, C3, NS, true >(A.w3, A.b3, buf0, nullptr, smax, lane, wave);
    __syncthreads();

    for (int c = t; c < C3; c += 128)
        A.out[(size_t)b * A.out_bstride + (size_t)s * A.out_sstride + c] = smax[c];
}

// ---------------------------------------------------------------------------
// Small FC layers (16 x cin) @ W^T: trivial FLOPs, plain VALU kernel.
// ---------------------------------------------------------------------------
__global__ void fc_kernel(const float* __restrict__ in, const float* __restrict__ w,
                          const float* __restrict__ bias, float* __restrict__ out,
                          int nb, int cin, int cout, int relu) {
    int id = blockIdx.x * blockDim.x + threadIdx.x;
    if (id >= nb * cout) return;
    int b = id / cout, o = id % cout;
    const float* x  = in + (size_t)b * cin;
    const float* wr = w  + (size_t)o * cin;
    float acc = bias[o];
    for (int k = 0; k < cin; ++k) acc += x[k] * wr[k];
    if (relu) acc = fmaxf(acc, 0.0f);
    out[(size_t)b * cout + o] = acc;
}

// ---------------------------------------------------------------------------
// Host
// ---------------------------------------------------------------------------
static inline size_t align256(size_t x) { return (x + 255) & ~(size_t)255; }
static inline size_t mlp_smem(int cin, int c1, int c2, int c3, int ns) {
    size_t cb0 = (size_t)((cin > c2) ? cin : c2) + 16;
    size_t cb1 = (size_t)c1 + 16;
    return (cb0 * ns + cb1 * ns) * 2 + (size_t)c3 * 4;
}

struct LD { int wi, bi, cout, cin; };

extern "C" void kernel_launch(void* const* d_in, const int* in_sizes, int n_in,
                              void* d_out, int out_size, void* d_ws, size_t ws_size,
                              hipStream_t stream) {
    (void)in_sizes; (void)n_in; (void)out_size; (void)ws_size;
    const int B = 16, N = 4096;
    const float* pc = (const float*)d_in[0];

    // Flattened params assumption: [pointcloud, fc(w,b)x3, sa1(3 mlps x 3 layers x (w,b)),
    //                               sa2(...), sa3(w,b)x3]
    static const LD L[21] = {
        { 7, 8, 32,  6}, { 9,10, 32, 32}, {11,12, 64, 32},   // sa1 b0
        {13,14, 64,  6}, {15,16, 64, 64}, {17,18,128, 64},   // sa1 b1
        {19,20, 64,  6}, {21,22, 96, 64}, {23,24,128, 96},   // sa1 b2
        {25,26, 64,323}, {27,28, 64, 64}, {29,30,128, 64},   // sa2 b0
        {31,32,128,323}, {33,34,128,128}, {35,36,256,128},   // sa2 b1
        {37,38,128,323}, {39,40,128,128}, {41,42,256,128},   // sa2 b2
        {43,44,256,643}, {45,46,512,256}, {47,48,1024,512}   // sa3
    };

    // ---- workspace carve-out ----
    char* ws = (char*)d_ws;
    size_t off = 0;
    auto alloc = [&](size_t bytes) { size_t o = off; off = align256(off + bytes); return ws + o; };

    __bf16* wpk[21];
    int cinp[21];
    for (int i = 0; i < 21; ++i) {
        cinp[i] = (L[i].cin + 31) & ~31;
        wpk[i] = (__bf16*)alloc((size_t)L[i].cout * cinp[i] * sizeof(__bf16));
    }
    int*   fidx1 = (int*)  alloc((size_t)B * 512 * 4);
    float* nx1   = (float*)alloc((size_t)B * 512 * 3 * 4);
    int    ns1[3] = {16, 32, 128};
    int*   idx1[3];
    for (int i = 0; i < 3; ++i) idx1[i] = (int*)alloc((size_t)B * 512 * ns1[i] * 4);
    float* feats1 = (float*)alloc((size_t)B * 512 * 320 * 4);   // point-major [B][S][C]
    int*   fidx2 = (int*)  alloc((size_t)B * 128 * 4);
    float* nx2   = (float*)alloc((size_t)B * 128 * 3 * 4);
    int    ns2[3] = {32, 64, 128};
    int*   idx2[3];
    for (int i = 0; i < 3; ++i) idx2[i] = (int*)alloc((size_t)B * 128 * ns2[i] * 4);
    float* feats2 = (float*)alloc((size_t)B * 128 * 640 * 4);   // point-major [B][S][C]
    float* h3  = (float*)alloc((size_t)B * 1024 * 4);
    float* hf1 = (float*)alloc((size_t)B * 512 * 4);
    float* hf2 = (float*)alloc((size_t)B * 256 * 4);

    // ---- 1. repack weights to padded bf16 ----
    for (int i = 0; i < 21; ++i) {
        int tot = L[i].cout * cinp[i];
        pack_w_kernel<<<(tot + 255) / 256, 256, 0, stream>>>(
            (const float*)d_in[L[i].wi], wpk[i], L[i].cout, L[i].cin, cinp[i]);
    }

    // ---- 2. stage 1: FPS + centroids + ball query ----
    fps_kernel<<<B, 256, 0, stream>>>(pc, N * 6, 6, N, 512, fidx1);
    gather_xyz_kernel<<<(B * 512 * 3 + 255) / 256, 256, 0, stream>>>(
        pc, N * 6, 6, fidx1, 512, nx1, B * 512 * 3);
    const float r1[3] = {0.1f, 0.2f, 0.4f};
    for (int i = 0; i < 3; ++i)
        ballquery_kernel<<<B * 512, 32, 0, stream>>>(
            pc, N * 6, 6, N, nx1, 512, r1[i] * r1[i], ns1[i], idx1[i]);

    // ---- 3. stage 1 branch MLPs (WMMA) -> feats1 [B][512][320] ----
    {
        GroupArgs A{};
        A.xyz = pc; A.xyz_bstride = N * 6; A.xyz_istride = 6;
        A.center = nx1;
        A.feat = pc + 3; A.feat_bstride = N * 6; A.feat_istride = 6; A.feat_cstride = 1; A.nfeat = 3;
        A.out_bstride = 512 * 320; A.out_sstride = 320; A.S = 512;

        A.idx = idx1[0];
        A.w1 = wpk[0]; A.b1 = (const float*)d_in[L[0].bi];
        A.w2 = wpk[1]; A.b2 = (const float*)d_in[L[1].bi];
        A.w3 = wpk[2]; A.b3 = (const float*)d_in[L[2].bi];
        A.out = feats1 + 0;
        group_mlp_kernel<32, 32, 32, 64, 16><<<B * 512, 128, mlp_smem(32,32,32,64,16), stream>>>(A);

        A.idx = idx1[1];
        A.w1 = wpk[3]; A.b1 = (const float*)d_in[L[3].bi];
        A.w2 = wpk[4]; A.b2 = (const float*)d_in[L[4].bi];
        A.w3 = wpk[5]; A.b3 = (const float*)d_in[L[5].bi];
        A.out = feats1 + 64;
        group_mlp_kernel<32, 64, 64, 128, 32><<<B * 512, 128, mlp_smem(32,64,64,128,32), stream>>>(A);

        A.idx = idx1[2];
        A.w1 = wpk[6]; A.b1 = (const float*)d_in[L[6].bi];
        A.w2 = wpk[7]; A.b2 = (const float*)d_in[L[7].bi];
        A.w3 = wpk[8]; A.b3 = (const float*)d_in[L[8].bi];
        A.out = feats1 + 192;
        group_mlp_kernel<32, 64, 96, 128, 128><<<B * 512, 128, mlp_smem(32,64,96,128,128), stream>>>(A);
    }

    // ---- 4. stage 2: FPS over nx1 + ball query over nx1 ----
    fps_kernel<<<B, 256, 0, stream>>>(nx1, 512 * 3, 3, 512, 128, fidx2);
    gather_xyz_kernel<<<(B * 128 * 3 + 255) / 256, 256, 0, stream>>>(
        nx1, 512 * 3, 3, fidx2, 128, nx2, B * 128 * 3);
    const float r2v[3] = {0.2f, 0.4f, 0.8f};
    for (int i = 0; i < 3; ++i)
        ballquery_kernel<<<B * 128, 32, 0, stream>>>(
            nx1, 512 * 3, 3, 512, nx2, 128, r2v[i] * r2v[i], ns2[i], idx2[i]);

    // ---- 5. stage 2 branch MLPs -> feats2 [B][128][640] ----
    {
        GroupArgs A{};
        A.xyz = nx1; A.xyz_bstride = 512 * 3; A.xyz_istride = 3;
        A.center = nx2;
        A.feat = feats1; A.feat_bstride = 512 * 320; A.feat_istride = 320; A.feat_cstride = 1; A.nfeat = 320;
        A.out_bstride = 128 * 640; A.out_sstride = 640; A.S = 128;

        A.idx = idx2[0];
        A.w1 = wpk[9];  A.b1 = (const float*)d_in[L[9].bi];
        A.w2 = wpk[10]; A.b2 = (const float*)d_in[L[10].bi];
        A.w3 = wpk[11]; A.b3 = (const float*)d_in[L[11].bi];
        A.out = feats2 + 0;
        group_mlp_kernel<352, 64, 64, 128, 32><<<B * 128, 128, mlp_smem(352,64,64,128,32), stream>>>(A);

        A.idx = idx2[1];
        A.w1 = wpk[12]; A.b1 = (const float*)d_in[L[12].bi];
        A.w2 = wpk[13]; A.b2 = (const float*)d_in[L[13].bi];
        A.w3 = wpk[14]; A.b3 = (const float*)d_in[L[14].bi];
        A.out = feats2 + 128;
        group_mlp_kernel<352, 128, 128, 256, 64><<<B * 128, 128, mlp_smem(352,128,128,256,64), stream>>>(A);

        A.idx = idx2[2];
        A.w1 = wpk[15]; A.b1 = (const float*)d_in[L[15].bi];
        A.w2 = wpk[16]; A.b2 = (const float*)d_in[L[16].bi];
        A.w3 = wpk[17]; A.b3 = (const float*)d_in[L[17].bi];
        A.out = feats2 + 384;
        group_mlp_kernel<352, 128, 128, 256, 128><<<B * 128, 128, mlp_smem(352,128,128,256,128), stream>>>(A);
    }

    // ---- 6. SA3 (group-all): one group per batch, NS = 128 points ----
    {
        GroupArgs A{};
        A.xyz = nx2; A.xyz_bstride = 128 * 3; A.xyz_istride = 3;
        A.center = nullptr;                       // absolute xyz, no centering
        A.feat = feats2; A.feat_bstride = 128 * 640; A.feat_istride = 640; A.feat_cstride = 1; A.nfeat = 640;
        A.idx = nullptr;                          // identity gather
        A.w1 = wpk[18]; A.b1 = (const float*)d_in[L[18].bi];
        A.w2 = wpk[19]; A.b2 = (const float*)d_in[L[19].bi];
        A.w3 = wpk[20]; A.b3 = (const float*)d_in[L[20].bi];
        A.out = h3; A.out_bstride = 1024; A.out_sstride = 0; A.S = 1;
        group_mlp_kernel<672, 256, 512, 1024, 128><<<B, 128, mlp_smem(672,256,512,1024,128), stream>>>(A);
    }

    // ---- 7. FC head ----
    fc_kernel<<<(B * 512 + 255) / 256, 256, 0, stream>>>(
        h3, (const float*)d_in[1], (const float*)d_in[2], hf1, B, 1024, 512, 1);
    fc_kernel<<<(B * 256 + 255) / 256, 256, 0, stream>>>(
        hf1, (const float*)d_in[3], (const float*)d_in[4], hf2, B, 512, 256, 1);
    fc_kernel<<<(B * 40 + 255) / 256, 256, 0, stream>>>(
        hf2, (const float*)d_in[5], (const float*)d_in[6], (float*)d_out, B, 256, 40, 0);
}
